// ColBERT_35725537968598
// MI455X (gfx1250) — matrix operand order repaired and checked by reading, not compiled, not run on hardware
//
#include <hip/hip_runtime.h>
#include <hip/hip_bf16.h>
#include <math.h>

#define PAD_TOKEN_ID 0
#define B_Q 64
#define L_Q 32
#define B_D 512
#define L_D 256
#define HID 768
#define OUTF 128
#define NSKIP 64

#define Q_TOK (B_Q * L_Q)          // 2048
#define D_TOK (B_D * L_D)          // 131072
#define QT (Q_TOK / 16)            // 128 query tiles
#define DT (D_TOK / 16)            // 8192 doc tiles
#define NKS (HID / 32)             // 24 K-steps
#define WAVES_PER_WG 16            // 16 waves -> 16 tiles per workgroup
#define NTILES (QT + DT)           // 8320 (multiple of 16)

// LDS: W staged in WMMA-fragment order: frag(ks,nt) = 32 lanes x 32 bytes
#define FRAG_BYTES 1024
#define NFRAGS (NKS * 8)                     // 192
#define WLDS_BYTES (NFRAGS * FRAG_BYTES)     // 196608 B < 320 KB WGP LDS

typedef __bf16 bf16_t;
typedef bf16_t v2bf  __attribute__((ext_vector_type(2)));
typedef bf16_t v8bf  __attribute__((ext_vector_type(8)));
typedef bf16_t v16bf __attribute__((ext_vector_type(16)));
typedef float  v8f   __attribute__((ext_vector_type(8)));

// packed f32x2 -> bf16x2 (RNE); native v_cvt_pk_bf16_f32 (confirmed on gfx1250)
static __device__ __forceinline__ v2bf pk2(float x, float y) {
#if __has_builtin(__builtin_amdgcn_cvt_pk_bf16_f32)
    return __builtin_amdgcn_cvt_pk_bf16_f32(x, y);
#else
    v2bf r; r[0] = (bf16_t)x; r[1] = (bf16_t)y; return r;
#endif
}

static __device__ __forceinline__ v16bf cat16(v8bf lo, v8bf hi) {
    return __builtin_shufflevector(lo, hi, 0,1,2,3,4,5,6,7,8,9,10,11,12,13,14,15);
}

union abuild16 { v16bf v; v2bf p[8]; };

// ---------------- Kernel 0: W f32 -> bf16 (once; Wb stays L2-resident) -----
__global__ __launch_bounds__(256)
void colbert_wcvt_kernel(const float* __restrict__ W, bf16_t* __restrict__ Wb) {
    int i = (blockIdx.x * 256 + threadIdx.x) * 2;
    if (i < OUTF * HID) {
        v2bf r = pk2(W[i], W[i + 1]);
        *(v2bf*)(Wb + i) = r;
    }
}

// ---------------- Kernel 1: projection + L2 norm + doc mask -----------------
// 512 threads = 16 waves; W (bf16) staged in LDS in fragment order once per
// workgroup; each wave runs a 16-token x 128-feature tile: 24 K-steps x 8
// N-tile WMMAs with double-buffered B-fragments from ds_load_b128 pairs.
__global__ __launch_bounds__(512)
void colbert_proj_kernel(const float* __restrict__ qh, const float* __restrict__ dh,
                         const int* __restrict__ ids, const int* __restrict__ skip,
                         const bf16_t* __restrict__ Wb,
                         bf16_t* __restrict__ Qn, bf16_t* __restrict__ Dn,
                         unsigned char* __restrict__ maskOut) {
    extern __shared__ char smem[];
    bf16_t* ldsW = (bf16_t*)smem;

    const int tid  = threadIdx.x;
    const int wave = tid >> 5;
    const int lane = tid & 31;
    const int lr   = lane & 15;
    const int half = lane >> 4;

    // ---- cooperative fill: Wb -> LDS in fragment order -------------------
    #pragma unroll
    for (int iter = 0; iter < (NFRAGS * 32) / 512; ++iter) {   // 12 iterations
        const int idx    = iter * 512 + tid;                   // 0..6143
        const int l32    = idx & 31;
        const int f      = idx >> 5;                           // 0..191
        const int nt     = f & 7;
        const int ks     = f >> 3;
        const int n      = nt * 16 + (l32 & 15);
        const int k      = ks * 32 + (l32 >> 4) * 16;
        const bf16_t* sp = Wb + (size_t)n * HID + k;
        bf16_t* dp       = ldsW + (size_t)idx * 16;
        *(v8bf*)(dp + 0) = *(const v8bf*)(sp + 0);
        *(v8bf*)(dp + 8) = *(const v8bf*)(sp + 8);
    }
    __syncthreads();

    // ---- per-wave tile ----------------------------------------------------
    const int tile = blockIdx.x * WAVES_PER_WG + wave;
    const bool isQ = tile < QT;
    int t0;
    const float* X;
    bf16_t* Yp;
    if (isQ) { t0 = tile * 16;        X = qh + (size_t)t0 * HID; Yp = Qn + (size_t)t0 * OUTF; }
    else     { t0 = (tile - QT) * 16; X = dh + (size_t)t0 * HID; Yp = Dn + (size_t)t0 * OUTF; }

    // mask for token t0+lr (doc tiles only); lanes 16..31 duplicate 0..15
    int mk = 1;
    if (!isQ) {
        const int id = ids[t0 + lr];
        mk = (id != PAD_TOKEN_ID) ? 1 : 0;
        #pragma unroll 8
        for (int j = 0; j < NSKIP; ++j) mk &= (id != skip[j]) ? 1 : 0;
        if (lane < 16) maskOut[t0 + lr] = (unsigned char)mk;
    }

    v8f acc[8];
    #pragma unroll
    for (int nt = 0; nt < 8; ++nt) {
        v8f z = {0.f,0.f,0.f,0.f,0.f,0.f,0.f,0.f};
        acc[nt] = z;
    }

    const float*  arow  = X + (size_t)lr * HID;
    const bf16_t* lbase = ldsW + (size_t)lane * 16;   // lane's slot in each frag

    #pragma unroll 1
    for (int ks = 0; ks < NKS; ++ks) {
        const int k0 = ks * 32;
        const float* ap = arow + k0 + half * 16;

        // prefetch next K-step's activation segment (global_prefetch_b8)
        if (ks + 1 < NKS) __builtin_prefetch(ap + 32, 0, 0);

        // A-frag (16x32 bf16): K kb..kb+15, kb = half*16
        const float4 a0 = *(const float4*)(ap + 0);
        const float4 a1 = *(const float4*)(ap + 4);
        const float4 a2 = *(const float4*)(ap + 8);
        const float4 a3 = *(const float4*)(ap + 12);

        abuild16 A;
        A.p[0] = pk2(a0.x, a0.y); A.p[1] = pk2(a0.z, a0.w);
        A.p[2] = pk2(a1.x, a1.y); A.p[3] = pk2(a1.z, a1.w);
        A.p[4] = pk2(a2.x, a2.y); A.p[5] = pk2(a2.z, a2.w);
        A.p[6] = pk2(a3.x, a3.y); A.p[7] = pk2(a3.z, a3.w);

        // double-buffered B-fragment pipeline: load frag nt+1 before WMMA nt
        const bf16_t* fb = lbase + (size_t)(ks * 8) * 512;
        v16bf Bcur = cat16(*(const v8bf*)(fb), *(const v8bf*)(fb + 8));
        #pragma unroll
        for (int nt = 0; nt < 8; ++nt) {
            v16bf Bnxt = Bcur;
            if (nt < 7) {
                const bf16_t* bp = fb + (size_t)(nt + 1) * 512;
                Bnxt = cat16(*(const v8bf*)(bp), *(const v8bf*)(bp + 8));
            }
            acc[nt] = __builtin_amdgcn_wmma_f32_16x16x32_bf16(
                false, A.v, false, Bcur, (short)0, acc[nt], false, false);
            Bcur = Bnxt;
        }
    }

    // L2 normalize per token row (C layout: vgpr r, half -> row m = r + half*8)
    #pragma unroll
    for (int r = 0; r < 8; ++r) {
        float s = 0.f;
        #pragma unroll
        for (int nt = 0; nt < 8; ++nt) { const float v = acc[nt][r]; s += v * v; }
        s += __shfl_xor(s, 1, 32);
        s += __shfl_xor(s, 2, 32);
        s += __shfl_xor(s, 4, 32);
        s += __shfl_xor(s, 8, 32);
        const float inv = 1.f / fmaxf(sqrtf(s), 1e-12f);
        const int m = r + half * 8;
        float scale = inv;
        if (!isQ) scale *= (float)__shfl(mk, m, 32);
        bf16_t* yrow = Yp + (size_t)m * OUTF;
        #pragma unroll
        for (int nt = 0; nt < 8; nt += 2) {
            v2bf pr = pk2(acc[nt][r] * scale, acc[nt + 1][r] * scale);
            yrow[(nt + 0) * 16 + lr] = pr[0];
            yrow[(nt + 1) * 16 + lr] = pr[1];
        }
    }
}

// ---------------- Kernel 2: MaxSim (Q_b 32x128 @ D_{b,n}^T 128x256) ---------
__global__ __launch_bounds__(32)
void colbert_maxsim_kernel(const bf16_t* __restrict__ Qn, const bf16_t* __restrict__ Dn,
                           const unsigned char* __restrict__ maskB,
                           float* __restrict__ out) {
    const int lane = threadIdx.x;
    const int lr   = lane & 15;
    const int half = lane >> 4;
    const int idx  = blockIdx.x;      // 0..511
    const int b    = idx >> 3;
    const int np   = idx & 7;

    const int qbase = b * L_Q;
    const int dbase = (b * 8 + np) * L_D;

    // hoist all Q fragments (2 M-tiles x 4 K-steps) into registers (64 VGPRs)
    v16bf Af[2][4];
    #pragma unroll
    for (int mt = 0; mt < 2; ++mt) {
        const bf16_t* qrow = Qn + (size_t)(qbase + mt * 16 + lr) * OUTF;
        #pragma unroll
        for (int ks = 0; ks < 4; ++ks) {
            const bf16_t* ap = qrow + ks * 32 + half * 16;
            Af[mt][ks] = cat16(*(const v8bf*)(ap), *(const v8bf*)(ap + 8));
        }
    }

    float maxv[2][8];
    #pragma unroll
    for (int mt = 0; mt < 2; ++mt)
        #pragma unroll
        for (int r = 0; r < 8; ++r) maxv[mt][r] = -INFINITY;

    // software pipeline: tile 0 preload
    v16bf Bf[4];
    int mcol;
    {
        const int dt = dbase + lr;
        mcol = maskB[dt];
        const bf16_t* drow = Dn + (size_t)dt * OUTF;
        __builtin_prefetch(drow + 16 * OUTF, 0, 0);
        #pragma unroll
        for (int ks = 0; ks < 4; ++ks) {
            const bf16_t* bp = drow + ks * 32 + half * 16;
            Bf[ks] = cat16(*(const v8bf*)(bp), *(const v8bf*)(bp + 8));
        }
    }

    #pragma unroll 1
    for (int ntile = 0; ntile < 16; ++ntile) {
        // preload next doc-tile's fragments + mask while this tile's WMMAs run
        v16bf Bn[4];
        int mnext = 0;
        if (ntile < 15) {
            const int dt2 = dbase + (ntile + 1) * 16 + lr;
            mnext = maskB[dt2];
            const bf16_t* drow2 = Dn + (size_t)dt2 * OUTF;
            if (ntile + 2 < 16) __builtin_prefetch(drow2 + 16 * OUTF, 0, 0);
            #pragma unroll
            for (int ks = 0; ks < 4; ++ks) {
                const bf16_t* bp = drow2 + ks * 32 + half * 16;
                Bn[ks] = cat16(*(const v8bf*)(bp), *(const v8bf*)(bp + 8));
            }
        }

        v8f c0 = {0.f,0.f,0.f,0.f,0.f,0.f,0.f,0.f};
        v8f c1 = c0;
        #pragma unroll
        for (int ks = 0; ks < 4; ++ks) {
            c0 = __builtin_amdgcn_wmma_f32_16x16x32_bf16(
                false, Af[0][ks], false, Bf[ks], (short)0, c0, false, false);
            c1 = __builtin_amdgcn_wmma_f32_16x16x32_bf16(
                false, Af[1][ks], false, Bf[ks], (short)0, c1, false, false);
        }

        // masked running max; one doc column per lane -> branchless select
        #pragma unroll
        for (int r = 0; r < 8; ++r) {
            const float v0 = mcol ? c0[r] : -INFINITY;
            const float v1 = mcol ? c1[r] : -INFINITY;
            maxv[0][r] = fmaxf(maxv[0][r], v0);
            maxv[1][r] = fmaxf(maxv[1][r], v1);
        }

        #pragma unroll
        for (int ks = 0; ks < 4; ++ks) Bf[ks] = Bn[ks];
        mcol = mnext;
    }

    // max-reduce each q-row across the 16 lanes of its half, then mean
    float s = 0.f;
    #pragma unroll
    for (int mt = 0; mt < 2; ++mt) {
        #pragma unroll
        for (int r = 0; r < 8; ++r) {
            float v = maxv[mt][r];
            v = fmaxf(v, __shfl_xor(v, 1, 32));
            v = fmaxf(v, __shfl_xor(v, 2, 32));
            v = fmaxf(v, __shfl_xor(v, 4, 32));
            v = fmaxf(v, __shfl_xor(v, 8, 32));
            s += v;
        }
    }
    s += __shfl_xor(s, 16, 32);
    if (lane == 0) out[idx] = s * (1.0f / 32.0f);
}

// ---------------------------------------------------------------------------
extern "C" void kernel_launch(void* const* d_in, const int* in_sizes, int n_in,
                              void* d_out, int out_size, void* d_ws, size_t ws_size,
                              hipStream_t stream) {
    const float* qh   = (const float*)d_in[0];
    const float* dh   = (const float*)d_in[1];
    const int*   ids  = (const int*)d_in[2];
    const int*   skip = (const int*)d_in[3];
    const float* W    = (const float*)d_in[4];
    float* out = (float*)d_out;

    // workspace layout
    const size_t QBYTES = (size_t)Q_TOK * OUTF * sizeof(bf16_t);   // 512 KB
    const size_t DBYTES = (size_t)D_TOK * OUTF * sizeof(bf16_t);   // 32 MB
    const size_t MBYTES = (size_t)D_TOK;                           // 128 KB
    bf16_t*        Qn   = (bf16_t*)d_ws;
    bf16_t*        Dn   = (bf16_t*)((char*)d_ws + QBYTES);
    unsigned char* mask = (unsigned char*)((char*)d_ws + QBYTES + DBYTES);
    bf16_t*        Wb   = (bf16_t*)((char*)d_ws + QBYTES + DBYTES + MBYTES);

    colbert_wcvt_kernel<<<(OUTF * HID / 2 + 255) / 256, 256, 0, stream>>>(W, Wb);
    colbert_proj_kernel<<<NTILES / WAVES_PER_WG, WAVES_PER_WG * 32, WLDS_BYTES, stream>>>(
        qh, dh, ids, skip, Wb, Qn, Dn, mask);
    colbert_maxsim_kernel<<<B_Q * 8, 32, 0, stream>>>(Qn, Dn, mask, out);
}
